// NeuralODE_31473520345331
// MI455X (gfx1250) — matrix operand order repaired and checked by reading, not compiled, not run on hardware
//
#include <hip/hip_runtime.h>
#include <hip/hip_bf16.h>

typedef unsigned short ushort_t;
typedef __attribute__((ext_vector_type(16))) __bf16 v16bf;
typedef __attribute__((ext_vector_type(8)))  float  v8f;

#define BSZ  512
#define TN   128
#define CDIM 512
#define HDIM 1024
#define MB   16            // batch rows per workgroup -> 32 workgroups
#define NTHR 512           // 16 waves
#define LDY  (CDIM + 8)    // bf16 row stride for yin   (row stride mod 64 banks = 4)
#define LDH  (HDIM + 8)    // bf16 row stride for h

union ABTile { uint4 q[2]; v16bf v; };

__device__ __forceinline__ ushort_t f2bf(float f) {
    unsigned u = __float_as_uint(f);
    unsigned r = (u + 0x7FFFu + ((u >> 16) & 1u)) >> 16;   // round-to-nearest-even
    return (ushort_t)r;
}

// tanh(x) = 1 - 2/(exp(2x)+1); v_exp_f32 + v_rcp_f32 (both TRANS, co-execute with VALU)
__device__ __forceinline__ float fast_tanh(float x) {
    float cx = fminf(fmaxf(x, -10.f), 10.f);
    float e  = __expf(2.f * cx);
    return 1.f - 2.f * __builtin_amdgcn_rcpf(e + 1.f);
}

__device__ __forceinline__ ABTile load_a(const ushort_t* base) {
    ABTile a;
    a.q[0] = *(const uint4*)base;
    a.q[1] = *(const uint4*)(base + 8);
    return a;
}
__device__ __forceinline__ ABTile load_b(const ushort_t* base) {
    ABTile b;
    b.q[0] = *(const uint4*)base;
    b.q[1] = *(const uint4*)(base + 8);
    return b;
}

// Pre-pass: W (R x Cc fp32, row-major)  ->  WT (Cc x R bf16, row-major over K=R)
__global__ void transpose_to_bf16(const float* __restrict__ W, ushort_t* __restrict__ WT,
                                  int R, int Cc) {
    int idx = blockIdx.x * 256 + threadIdx.x;
    if (idx < R * Cc) {
        int r = idx / Cc, c = idx % Cc;
        WT[(size_t)c * R + r] = f2bf(W[idx]);
    }
}

__global__ __launch_bounds__(NTHR) void node_rk4_kernel(
    const float* __restrict__ x,
    const float* __restrict__ b1, const float* __restrict__ b2, const float* __restrict__ b3,
    const ushort_t* __restrict__ W1T,   // 1024 x 512
    const ushort_t* __restrict__ W2T,   // 1024 x 1024
    const ushort_t* __restrict__ W3T,   //  512 x 1024
    float* __restrict__ out)
{
    __shared__ ushort_t s_yin[MB * LDY];   // bf16 input to GEMM1        (16.6 KB)
    __shared__ ushort_t s_h  [MB * LDH];   // bf16 h1 then h2 (reused)   (33.0 KB)

    const int b0   = blockIdx.x * MB;
    const int tid  = threadIdx.x;
    const int w    = tid >> 5;        // wave id 0..15 (wave32)
    const int lane = tid & 31;
    const int lm   = lane & 15;       // WMMA row(A)/col(B,D) selector
    const int hi   = lane >> 4;       // K-half selector
    const float dt = 1.f / (float)(TN - 1);

    // Persistent RK4 state in WMMA D layout: tile i covers cols (w+16i)*16..+15 of C
    float yreg[2][8];
    float accr[2][8];

    // ---- init: y0 = x[:,0,:]; emit out[:,0,:]; seed yin ----
    for (int i = tid; i < MB * CDIM; i += NTHR) {
        int m = i >> 9, c = i & (CDIM - 1);
        float v = x[(size_t)(b0 + m) * TN * CDIM + c];
        out[(size_t)(b0 + m) * TN * CDIM + c] = v;
        s_yin[m * LDY + c] = f2bf(v);
    }
#pragma unroll
    for (int i = 0; i < 2; ++i) {
        int n = (w + 16 * i) * 16 + lm;
#pragma unroll
        for (int v = 0; v < 8; ++v) {
            int m = v + 8 * hi;
            yreg[i][v] = x[(size_t)(b0 + m) * TN * CDIM + n];
        }
    }
    __syncthreads();

    const int aoff_y = lm * LDY + 16 * hi;   // A-operand lane base (yin)
    const int aoff_h = lm * LDH + 16 * hi;   // A-operand lane base (h)

    for (int t = 1; t < TN; ++t) {
        for (int s = 0; s < 4; ++s) {
            // ---------- GEMM1: h1 = tanh(yin @ W1 + b1), N=1024, K=512 ----------
            {
                v8f c[4];
#pragma unroll
                for (int i = 0; i < 4; ++i) c[i] = (v8f){0.f,0.f,0.f,0.f,0.f,0.f,0.f,0.f};
#pragma unroll 2
                for (int k = 0; k < CDIM; k += 32) {
                    ABTile a = load_a(s_yin + aoff_y + k);
#pragma unroll
                    for (int i = 0; i < 4; ++i) {
                        int n = (w + 16 * i) * 16 + lm;
                        ABTile b = load_b(W1T + (size_t)n * CDIM + k + 16 * hi);
                        c[i] = __builtin_amdgcn_wmma_f32_16x16x32_bf16(
                                   false, a.v, false, b.v, (short)0, c[i], false, false);
                    }
                }
#pragma unroll
                for (int i = 0; i < 4; ++i) {
                    int n = (w + 16 * i) * 16 + lm;
                    float bias = b1[n];
#pragma unroll
                    for (int v = 0; v < 8; ++v) {
                        int m = v + 8 * hi;
                        s_h[m * LDH + n] = f2bf(fast_tanh(c[i][v] + bias));
                    }
                }
            }
            __syncthreads();

            // ---------- GEMM2: h2 = tanh(h1 @ W2 + b2), accumulators double as staging ----
            {
                v8f c[4];
#pragma unroll
                for (int i = 0; i < 4; ++i) c[i] = (v8f){0.f,0.f,0.f,0.f,0.f,0.f,0.f,0.f};
#pragma unroll 2
                for (int k = 0; k < HDIM; k += 32) {
                    ABTile a = load_a(s_h + aoff_h + k);
#pragma unroll
                    for (int i = 0; i < 4; ++i) {
                        int n = (w + 16 * i) * 16 + lm;
                        ABTile b = load_b(W2T + (size_t)n * HDIM + k + 16 * hi);
                        c[i] = __builtin_amdgcn_wmma_f32_16x16x32_bf16(
                                   false, a.v, false, b.v, (short)0, c[i], false, false);
                    }
                }
                __syncthreads();   // all h1 reads done -> safe to overwrite with h2
#pragma unroll
                for (int i = 0; i < 4; ++i) {
                    int n = (w + 16 * i) * 16 + lm;
                    float bias = b2[n];
#pragma unroll
                    for (int v = 0; v < 8; ++v) {
                        int m = v + 8 * hi;
                        s_h[m * LDH + n] = f2bf(fast_tanh(c[i][v] + bias));
                    }
                }
            }
            __syncthreads();

            // ---------- GEMM3: k = h2 @ W3 + b3, fused RK4 combine ----------
            {
                v8f c[2];
#pragma unroll
                for (int i = 0; i < 2; ++i) c[i] = (v8f){0.f,0.f,0.f,0.f,0.f,0.f,0.f,0.f};
#pragma unroll 2
                for (int k = 0; k < HDIM; k += 32) {
                    ABTile a = load_a(s_h + aoff_h + k);
#pragma unroll
                    for (int i = 0; i < 2; ++i) {
                        int n = (w + 16 * i) * 16 + lm;
                        ABTile b = load_b(W3T + (size_t)n * HDIM + k + 16 * hi);
                        c[i] = __builtin_amdgcn_wmma_f32_16x16x32_bf16(
                                   false, a.v, false, b.v, (short)0, c[i], false, false);
                    }
                }
#pragma unroll
                for (int i = 0; i < 2; ++i) {
                    int n = (w + 16 * i) * 16 + lm;
                    float bias = b3[n];
#pragma unroll
                    for (int v = 0; v < 8; ++v) {
                        float kk = c[i][v] + bias;
                        float yv = yreg[i][v];
                        float yin;
                        int m = v + 8 * hi;
                        if (s == 0)      { accr[i][v]  = kk;        yin = yv + 0.5f * dt * kk; }
                        else if (s == 1) { accr[i][v] += 2.f * kk;  yin = yv + 0.5f * dt * kk; }
                        else if (s == 2) { accr[i][v] += 2.f * kk;  yin = yv + dt * kk; }
                        else {
                            yv += (dt / 6.f) * (accr[i][v] + kk);
                            yreg[i][v] = yv;
                            yin = yv;
                            out[(size_t)(b0 + m) * TN * CDIM + (size_t)t * CDIM + n] = yv;
                        }
                        s_yin[m * LDY + n] = f2bf(yin);
                    }
                }
            }
            __syncthreads();
        }
    }
}

extern "C" void kernel_launch(void* const* d_in, const int* in_sizes, int n_in,
                              void* d_out, int out_size, void* d_ws, size_t ws_size,
                              hipStream_t stream) {
    (void)in_sizes; (void)n_in; (void)out_size; (void)ws_size;
    const float* x  = (const float*)d_in[0];
    const float* W1 = (const float*)d_in[1];
    const float* b1 = (const float*)d_in[2];
    const float* W2 = (const float*)d_in[3];
    const float* b2 = (const float*)d_in[4];
    const float* W3 = (const float*)d_in[5];
    const float* b3 = (const float*)d_in[6];
    float* out = (float*)d_out;

    ushort_t* W1T = (ushort_t*)d_ws;            // 1024x512  bf16 (1 MB)
    ushort_t* W2T = W1T + (size_t)CDIM * HDIM;  // 1024x1024 bf16 (2 MB)
    ushort_t* W3T = W2T + (size_t)HDIM * HDIM;  //  512x1024 bf16 (1 MB)

    transpose_to_bf16<<<(CDIM * HDIM + 255) / 256, 256, 0, stream>>>(W1, W1T, CDIM, HDIM);
    transpose_to_bf16<<<(HDIM * HDIM + 255) / 256, 256, 0, stream>>>(W2, W2T, HDIM, HDIM);
    transpose_to_bf16<<<(HDIM * CDIM + 255) / 256, 256, 0, stream>>>(W3, W3T, HDIM, CDIM);

    node_rk4_kernel<<<BSZ / MB, NTHR, 0, stream>>>(x, b1, b2, b3, W1T, W2T, W3T, out);
}
